// MultiHeadAttention_73246372266088
// MI455X (gfx1250) — compile-verified, matrix operand order checked by reading, not run on hardware
//
#include <hip/hip_runtime.h>
#include <hip/hip_bf16.h>
#include <stdint.h>

typedef __attribute__((ext_vector_type(16))) __bf16 v16bf;
typedef __attribute__((ext_vector_type(8)))  float  v8f;

#define BM 256
#define BN 128
#define BK 32
#define GEMM_THREADS 256
// async b128 instructions issued per wave per stage: (BM*4 + BN*4)/256 = 6
//   4 for the A tile, 2 for the B tile

static __device__ __forceinline__ unsigned short f2bf(float f) {
  unsigned int u = __builtin_bit_cast(unsigned int, f);
  u += 0x7FFFu + ((u >> 16) & 1u);          // round-to-nearest-even
  return (unsigned short)(u >> 16);
}

union Frag16 { uint4 u[2]; v16bf v; };

static __device__ __forceinline__ void async_b128(const void* gsrc, void* ldst) {
  asm volatile("global_load_async_to_lds_b128 %0, %1, off"
               :: "v"((unsigned)(unsigned long long)ldst),
                  "v"((unsigned long long)gsrc)
               : "memory");
}

// Wait until at most the newest in-flight stage (6 ops) remains.
static __device__ __forceinline__ void wait_async_stage() {
  asm volatile("s_wait_asynccnt 6" ::: "memory");
}

// ---------------------------------------------------------------------------
// Generic bf16 WMMA GEMM:  C = A(MxK, row-major) * Bt(NxK, N-major)^T
// Block tile 256x128, 8 wave32 in a 4x2 grid, 64x64 per wave (4x4 fragments).
// Triple-buffered LDS fed by global_load_async_to_lds_b128 (2 tiles in flight).
// mode 0: store bf16 row-major (batched via strideC)
// mode 1: store bf16 V-transposed per 512-row block: C[(m/512)*1024+n][m%512]
// mode 2: store f32 * scale row-major (batched)   -> attention scores
// mode 3: store f32 + bias[n] row-major           -> final projection
// ---------------------------------------------------------------------------
__global__ __launch_bounds__(GEMM_THREADS)
void gemm_bf16_wmma(const unsigned short* __restrict__ A,
                    const unsigned short* __restrict__ Bt,
                    void* __restrict__ Cv,
                    const float* __restrict__ bias,
                    int M, int N, int K,
                    long long strideA, long long strideB, long long strideC,
                    int mode, float scale)
{
  __shared__ __align__(16) unsigned short sA[3][BM * BK];   // 3 x 16 KB
  __shared__ __align__(16) unsigned short sB[3][BN * BK];   // 3 x  8 KB

  const int tid  = threadIdx.x;
  const int lane = tid & 31;
  const int wid  = tid >> 5;
  const int wm   = wid & 3;    // 0..3  (64-row strip of 256)
  const int wn   = wid >> 2;   // 0..1  (64-col strip of 128)
  const int bz   = blockIdx.z;

  const unsigned short* Ab = A  + (size_t)bz * strideA + (size_t)blockIdx.y * BM * K;
  const unsigned short* Bb = Bt + (size_t)bz * strideB + (size_t)blockIdx.x * BN * K;

  const int ktiles = K / BK;

  // Stage tile kt into buffer buf: 6 async b128 per wave (4 A, 2 B).
  auto stage = [&](int buf, int kt) {
    const unsigned short* ga = Ab + kt * BK;
    const unsigned short* gb = Bb + kt * BK;
#pragma unroll
    for (int i = 0; i < 4; ++i) {                 // A: 256 rows x 4 chunks
      const int c   = tid + i * GEMM_THREADS;     // 0..1023
      const int row = c >> 2;
      const int col = (c & 3) * 8;                // halfs; 16B chunks
      async_b128(ga + (size_t)row * K + col, &sA[buf][row * BK + col]);
    }
#pragma unroll
    for (int i = 0; i < 2; ++i) {                 // B: 128 rows x 4 chunks
      const int c   = tid + i * GEMM_THREADS;     // 0..511
      const int row = c >> 2;
      const int col = (c & 3) * 8;
      async_b128(gb + (size_t)row * K + col, &sB[buf][row * BK + col]);
    }
  };

  v8f acc[4][4] = {};

  const int lrow = lane & 15;
  const int kb   = (lane >> 4) * 8;   // ISA 16-bit A/B lane->K mapping

  stage(0, 0);
  if (ktiles > 1) stage(1, 1);

  for (int kt = 0; kt < ktiles; ++kt) {
    const int cur = kt % 3;

    wait_async_stage();   // stage kt landed (stage kt+1 may still be in flight)
    __syncthreads();      // ..for every wave; also: all waves done with buf cur+... (kt-1)

    if (kt + 2 < ktiles) stage((kt + 2) % 3, kt + 2);  // overwrites buf of kt-1: safe

    Frag16 af[4], bfr[4];
#pragma unroll
    for (int mi = 0; mi < 4; ++mi) {
      const unsigned short* p = &sA[cur][(wm * 64 + mi * 16 + lrow) * BK + kb];
      af[mi].u[0] = *(const uint4*)p;
      af[mi].u[1] = *(const uint4*)(p + 16);
    }
#pragma unroll
    for (int ni = 0; ni < 4; ++ni) {
      const unsigned short* p = &sB[cur][(wn * 64 + ni * 16 + lrow) * BK + kb];
      bfr[ni].u[0] = *(const uint4*)p;
      bfr[ni].u[1] = *(const uint4*)(p + 16);
    }
#pragma unroll
    for (int mi = 0; mi < 4; ++mi)
#pragma unroll
      for (int ni = 0; ni < 4; ++ni)
        acc[mi][ni] = __builtin_amdgcn_wmma_f32_16x16x32_bf16(
            false, af[mi].v, false, bfr[ni].v,
            (short)0, acc[mi][ni], false, false);

    __syncthreads();      // all waves done reading buf cur before it is restaged
  }

  // Epilogue: C/D 16x16 f32 layout: lane n = lane&15, VGPR r -> m = r + (lane>=16 ? 8 : 0)
  const int mrow_off = (lane < 16) ? 0 : 8;
#pragma unroll
  for (int mi = 0; mi < 4; ++mi) {
#pragma unroll
    for (int ni = 0; ni < 4; ++ni) {
      const int n = blockIdx.x * BN + wn * 64 + ni * 16 + lrow;
      const float bv = (mode == 3) ? bias[n] : 0.0f;
#pragma unroll
      for (int r = 0; r < 8; ++r) {
        const int m = blockIdx.y * BM + wm * 64 + mi * 16 + mrow_off + r;
        const float val = acc[mi][ni][r];
        if (mode == 0) {
          unsigned short* C = (unsigned short*)Cv + (size_t)bz * strideC;
          C[(size_t)m * N + n] = f2bf(val);
        } else if (mode == 1) {
          unsigned short* C = (unsigned short*)Cv;
          C[((size_t)(m >> 9) * 1024 + n) * 512 + (m & 511)] = f2bf(val);
        } else if (mode == 2) {
          float* C = (float*)Cv + (size_t)bz * strideC;
          C[(size_t)m * N + n] = val * scale;
        } else {
          float* C = (float*)Cv;
          C[(size_t)m * N + n] = val + bv;
        }
      }
    }
  }
}

// ---------------------------------------------------------------------------
// fp32 -> bf16 conversion (4 elems/thread, exact sizing)
// ---------------------------------------------------------------------------
__global__ void f32_to_bf16(const float* __restrict__ src, unsigned short* __restrict__ dst) {
  const size_t i = ((size_t)blockIdx.x * 256 + threadIdx.x) * 4;
  const float4 f = *(const float4*)(src + i);
  uint2 o;
  o.x = (unsigned)f2bf(f.x) | ((unsigned)f2bf(f.y) << 16);
  o.y = (unsigned)f2bf(f.z) | ((unsigned)f2bf(f.w) << 16);
  *(uint2*)(dst + i) = o;
}

// fp32 -> bf16 with 1024x1024 transpose (weights -> N-major B operand)
__global__ void f32_to_bf16_T(const float* __restrict__ src, unsigned short* __restrict__ dst) {
  const int idx = blockIdx.x * 256 + threadIdx.x;
  const int r = idx >> 10, c = idx & 1023;
  dst[(size_t)c * 1024 + r] = f2bf(src[idx]);
}

// ---------------------------------------------------------------------------
// In-place row softmax over 512-wide rows; also emit bf16 probs for ctx GEMM
// ---------------------------------------------------------------------------
__global__ void softmax_rows(float* __restrict__ sc, unsigned short* __restrict__ pb) {
  __shared__ float red[256];
  const size_t row = blockIdx.x;
  float* p = sc + row * 512;
  const int t = threadIdx.x;
  const float a = p[t], b = p[t + 256];
  red[t] = fmaxf(a, b);
  __syncthreads();
  for (int s = 128; s > 0; s >>= 1) {
    if (t < s) red[t] = fmaxf(red[t], red[t + s]);
    __syncthreads();
  }
  const float mx = red[0];
  __syncthreads();
  const float ea = __expf(a - mx), eb = __expf(b - mx);
  red[t] = ea + eb;
  __syncthreads();
  for (int s = 128; s > 0; s >>= 1) {
    if (t < s) red[t] += red[t + s];
    __syncthreads();
  }
  const float inv = 1.0f / red[0];
  const float ra = ea * inv, rb = eb * inv;
  p[t] = ra;  p[t + 256] = rb;
  pb[row * 512 + t] = f2bf(ra);
  pb[row * 512 + t + 256] = f2bf(rb);
}

// ---------------------------------------------------------------------------
// out = LayerNorm(residual + x) * gamma + beta, rows of 1024
// ---------------------------------------------------------------------------
__global__ void resid_ln(const float* __restrict__ xo, const float* __restrict__ xr,
                         const float* __restrict__ gamma, const float* __restrict__ beta,
                         float* __restrict__ out) {
  __shared__ float r1[256], r2[256];
  const size_t row = blockIdx.x;
  const int t = threadIdx.x;
  const float* po = xo + row * 1024;
  const float* pr = xr + row * 1024;
  float v[4], s = 0.f, sq = 0.f;
#pragma unroll
  for (int i = 0; i < 4; ++i) {
    const int j = t + i * 256;
    const float x = po[j] + pr[j];
    v[i] = x; s += x; sq += x * x;
  }
  r1[t] = s; r2[t] = sq;
  __syncthreads();
  for (int st = 128; st > 0; st >>= 1) {
    if (t < st) { r1[t] += r1[t + st]; r2[t] += r2[t + st]; }
    __syncthreads();
  }
  const float mean = r1[0] * (1.0f / 1024.0f);
  const float var  = r2[0] * (1.0f / 1024.0f) - mean * mean;
  const float inv  = rsqrtf(var + 1e-5f);
  float* po2 = out + row * 1024;
#pragma unroll
  for (int i = 0; i < 4; ++i) {
    const int j = t + i * 256;
    po2[j] = (v[i] - mean) * inv * gamma[j] + beta[j];
  }
}

// ---------------------------------------------------------------------------
extern "C" void kernel_launch(void* const* d_in, const int* in_sizes, int n_in,
                              void* d_out, int out_size, void* d_ws, size_t ws_size,
                              hipStream_t stream) {
  (void)in_sizes; (void)n_in; (void)out_size; (void)ws_size;

  const float* query = (const float*)d_in[0];
  const float* key   = (const float*)d_in[1];
  const float* value = (const float*)d_in[2];
  const float* wq    = (const float*)d_in[3];
  const float* wk    = (const float*)d_in[4];
  const float* wv    = (const float*)d_in[5];
  const float* wfin  = (const float*)d_in[6];
  const float* bfin  = (const float*)d_in[7];
  const float* gamma = (const float*)d_in[8];
  const float* beta  = (const float*)d_in[9];

  constexpr int D = 1024, H = 16;
  constexpr long long MS  = 4LL * 8192;      // B*S = 32768 rows
  constexpr long long ACT = MS * D;          // 33,554,432 elems
  constexpr int SB = 8192 / H;               // 512 rows per attention block
  constexpr int NBLK = 4 * H;                // 64 blocks
  const float scale = 0.125f;                // (D/H)^-0.5 = 64^-0.5

  // ---- workspace layout (bf16 unless noted) ----
  unsigned short* qin = (unsigned short*)d_ws;
  unsigned short* kin = qin + ACT;
  unsigned short* vin = kin + ACT;
  unsigned short* wqT = vin + ACT;
  unsigned short* wkT = wqT + (size_t)D * D;
  unsigned short* wvT = wkT + (size_t)D * D;
  unsigned short* wfB = wvT + (size_t)D * D;          // w_final as-is == (w_final^T) N-major
  unsigned short* Qb  = wfB + (size_t)D * D;
  unsigned short* Kb  = Qb + ACT;
  unsigned short* vT  = Kb + ACT;                      // [blk][n:1024][k:512]
  unsigned short* attnb = vT + ACT;                    // bf16 probs
  unsigned short* ctx = vin;                           // reuse: vin dead after V proj
  float* outF = (float*)qin;                           // reuse: qin+kin dead after projs

  float* out_main = (float*)d_out;                     // [32768,1024] f32
  float* attn_out = out_main + ACT;                    // [64,512,512]  f32 (in-place softmax)

  const dim3 blk(GEMM_THREADS);

  // 1) conversions
  f32_to_bf16<<<(unsigned)(ACT / 1024), 256, 0, stream>>>(query, qin);
  f32_to_bf16<<<(unsigned)(ACT / 1024), 256, 0, stream>>>(key,   kin);
  f32_to_bf16<<<(unsigned)(ACT / 1024), 256, 0, stream>>>(value, vin);
  f32_to_bf16_T<<<(D * D) / 256, 256, 0, stream>>>(wq, wqT);
  f32_to_bf16_T<<<(D * D) / 256, 256, 0, stream>>>(wk, wkT);
  f32_to_bf16_T<<<(D * D) / 256, 256, 0, stream>>>(wv, wvT);
  f32_to_bf16<<<(D * D) / 1024, 256, 0, stream>>>(wfin, wfB);

  // 2) projections  [32768x1024] = [32768x1024] x [1024x1024]
  const dim3 gp(D / BN, (unsigned)(MS / BM), 1);
  gemm_bf16_wmma<<<gp, blk, 0, stream>>>(qin, wqT, Qb, nullptr,
      (int)MS, D, D, 0, 0, 0, 0, 0.f);
  gemm_bf16_wmma<<<gp, blk, 0, stream>>>(kin, wkT, Kb, nullptr,
      (int)MS, D, D, 0, 0, 0, 0, 0.f);
  gemm_bf16_wmma<<<gp, blk, 0, stream>>>(vin, wvT, vT, nullptr,
      (int)MS, D, D, 0, 0, 0, 1, 0.f);   // V stored transposed per block

  // 3) scores[b] = Q[b] * K[b]^T * scale  (K rows serve as N-major B directly)
  const dim3 gs(SB / BN, SB / BM, NBLK);
  gemm_bf16_wmma<<<gs, blk, 0, stream>>>(Qb, Kb, attn_out, nullptr,
      SB, SB, D, (long long)SB * D, (long long)SB * D, (long long)SB * SB, 2, scale);

  // 4) softmax rows (in-place on d_out attention region) + bf16 copy
  softmax_rows<<<(unsigned)(NBLK * (long long)SB), 256, 0, stream>>>(attn_out, attnb);

  // 5) context[b] = attn[b] * V[b]   (vT is N-major B)
  const dim3 gc(D / BN, SB / BM, NBLK);
  gemm_bf16_wmma<<<gc, blk, 0, stream>>>(attnb, vT, ctx, nullptr,
      SB, D, SB, (long long)SB * SB, (long long)D * SB, (long long)SB * D, 0, 0.f);

  // 6) final = context * w_final^T + b_final (f32)
  gemm_bf16_wmma<<<gp, blk, 0, stream>>>(ctx, wfB, outF, bfin,
      (int)MS, D, D, 0, 0, 0, 3, 0.f);

  // 7) layernorm(residual + final)
  resid_ln<<<(unsigned)MS, 256, 0, stream>>>(outF, query, gamma, beta, out_main);
}
